// GCAN_60945585930897
// MI455X (gfx1250) — matrix operand order repaired and checked by reading, not compiled, not run on hardware
//
#include <hip/hip_runtime.h>
#include <hip/hip_bf16.h>

// ---------------- dims ----------------
#define Bn    64
#define Sn    128
#define Ln    512
#define DIN   12
#define GHID  256
#define GOUT  128
#define SIND  3347
#define SMID  256
#define SHID  32
#define FSZ   3
#define KSZ   64
#define PHID  64
#define COn   64
#define Pn    (Ln - FSZ)      // 509
#define FEAT  320             // v1(32)+q1(128)+v2(32)+q2(64)+gru2(64)

typedef __bf16 bf16_t;
typedef bf16_t v16bf __attribute__((ext_vector_type(16)));
typedef float  v8f   __attribute__((ext_vector_type(8)));

__device__ __forceinline__ float sigmoidf_(float x) { return 1.0f / (1.0f + expf(-x)); }

// ============================================================
// bf16 WMMA GEMM with N-blocking:
//   C[M,N] = act(A[M,K] * W[N,K]^T + bias)
// One wave computes a 16 x (16*NT) strip: the A fragment is loaded
// once per K-step and reused for NT consecutive B tiles (cuts A
// re-read traffic NT-fold vs one-tile-per-wave).
// M multiple of 16, N multiple of 16*NT, K arbitrary. act: 0=none,1=tanh
// ============================================================
template <int NT>
__global__ __launch_bounds__(32)
void wmma_gemm_nt(const float* __restrict__ A,
                  const float* __restrict__ W,
                  const float* __restrict__ bias,
                  float* __restrict__ C,
                  int M, int N, int K, int act) {
  const int lane = threadIdx.x & 31;
  const int half = lane >> 4;          // which half-wave
  const int l    = lane & 15;
  const int m0   = blockIdx.x << 4;
  const int n0   = blockIdx.y * (16 * NT);

  const float* __restrict__ Arow = A + (size_t)(m0 + l) * K;
  const float* __restrict__ Wrow = W + (size_t)(n0 + l) * K;   // tile nt: + nt*16*K

  v8f acc[NT];
#pragma unroll
  for (int nt = 0; nt < NT; ++nt) acc[nt] = (v8f){};

  const int Kfull = K & ~31;
  int k0 = 0;
  for (; k0 < Kfull; k0 += 32) {
    // A 16x32 bf16 fragment (ISA 7.12.2): half 0 -> K=[k0..k0+7],[k0+16..k0+23]
    //                                     half 1 -> K=[k0+8..k0+15],[k0+24..k0+31]
    const float* ap0 = Arow + k0 + half * 8;
    const float* ap1 = Arow + k0 + 16 + half * 8;
    __builtin_prefetch(Arow + k0 + 128, 0, 0);
    v16bf a;
#pragma unroll
    for (int i = 0; i < 8; ++i) {
      a[i]     = (bf16_t)ap0[i];
      a[i + 8] = (bf16_t)ap1[i];
    }
#pragma unroll
    for (int nt = 0; nt < NT; ++nt) {
      // B 32x16 bf16 fragment: half 0 -> K=[k0..k0+7],[k0+16..k0+23]; half 1 -> K=[k0+8..k0+15],[k0+24..k0+31]
      const float* bp = Wrow + (size_t)nt * 16 * K + k0 + half * 8;
      v16bf bfr;
#pragma unroll
      for (int i = 0; i < 16; ++i) bfr[i] = (bf16_t)bp[i < 8 ? i : i + 8];
      acc[nt] = __builtin_amdgcn_wmma_f32_16x16x32_bf16(false, a, false, bfr,
                                                        (short)0, acc[nt], false, false);
    }
  }
  if (k0 < K) {   // guarded K tail (runs at most once)
    v16bf a;
#pragma unroll
    for (int i = 0; i < 8; ++i) {
      const int ka0 = k0 + half * 8 + i;
      const int ka1 = k0 + 16 + half * 8 + i;
      a[i]     = (ka0 < K) ? (bf16_t)Arow[ka0] : (bf16_t)0.0f;
      a[i + 8] = (ka1 < K) ? (bf16_t)Arow[ka1] : (bf16_t)0.0f;
    }
#pragma unroll
    for (int nt = 0; nt < NT; ++nt) {
      const float* bp = Wrow + (size_t)nt * 16 * K;
      v16bf bfr;
#pragma unroll
      for (int i = 0; i < 16; ++i) {
        const int kb = k0 + half * 8 + (i < 8 ? i : i + 8);
        bfr[i] = (kb < K) ? (bf16_t)bp[kb] : (bf16_t)0.0f;
      }
      acc[nt] = __builtin_amdgcn_wmma_f32_16x16x32_bf16(false, a, false, bfr,
                                                        (short)0, acc[nt], false, false);
    }
  }
  // D 16x16 f32: VGPR r -> row r + 8*half, col = lane&15
#pragma unroll
  for (int nt = 0; nt < NT; ++nt) {
    const int col = n0 + nt * 16 + l;
    const float bv = bias ? bias[col] : 0.0f;
#pragma unroll
    for (int r = 0; r < 8; ++r) {
      const int row = m0 + r + half * 8;
      float v = acc[nt][r] + bv;
      if (act == 1) v = tanhf(v);
      C[(size_t)row * N + col] = v;
    }
  }
}

// ============================================================
// GCN: g[b,:] = mean_l(X) W1 W2 + biases (mean commutes with linear)
// ============================================================
__global__ __launch_bounds__(GHID)
void gcn_kernel(const float* __restrict__ X, const float* __restrict__ W1,
                const float* __restrict__ b1, const float* __restrict__ W2,
                const float* __restrict__ b2, float* __restrict__ g) {
  const int b = blockIdx.x, t = threadIdx.x;
  __shared__ float mx[DIN];
  __shared__ float h1s[GHID];
  if (t < DIN) {
    float s = 0.0f;
    const float* xb = X + (size_t)b * Ln * DIN + t;
    for (int l2 = 0; l2 < Ln; ++l2) s += xb[l2 * DIN];
    mx[t] = s * (1.0f / (float)Ln);
  }
  __syncthreads();
  {
    float acc = b1[t];
#pragma unroll
    for (int d = 0; d < DIN; ++d) acc += mx[d] * W1[d * GHID + t];
    h1s[t] = acc;
  }
  __syncthreads();
  if (t < GOUT) {
    float acc = b2[t];
    for (int j = 0; j < GHID; ++j) acc += h1s[j] * W2[j * GOUT + t];
    g[b * GOUT + t] = acc;
  }
}

// ============================================================
// Source GRU: hidden 32, 128 steps; xp precomputed (incl. b_ih).
// One block per batch, 96 threads, Whh cached in LDS.
// ============================================================
__global__ __launch_bounds__(3 * SHID)
void src_gru_kernel(const float* __restrict__ xp, const float* __restrict__ Whh,
                    const float* __restrict__ bhh, float* __restrict__ out) {
  const int b = blockIdx.x, j = threadIdx.x;   // 96 threads
  __shared__ float Wsh[3 * SHID * SHID];
  __shared__ float h[SHID];
  __shared__ float gh[3 * SHID];
  for (int i = j; i < 3 * SHID * SHID; i += 3 * SHID) Wsh[i] = Whh[i];
  if (j < SHID) h[j] = 0.0f;
  const float bj = bhh[j];
  __syncthreads();
  for (int t = 0; t < Sn; ++t) {
    float acc = bj;
    const float* wr = Wsh + j * SHID;
#pragma unroll
    for (int k = 0; k < SHID; ++k) acc += h[k] * wr[k];
    gh[j] = acc;
    __syncthreads();
    if (j < SHID) {
      const float* xrow = xp + (size_t)(b * Sn + t) * (3 * SHID);
      const float r = sigmoidf_(xrow[j] + gh[j]);
      const float z = sigmoidf_(xrow[SHID + j] + gh[SHID + j]);
      const float n = tanhf(xrow[2 * SHID + j] + r * gh[2 * SHID + j]);
      const float h2 = (1.0f - z) * n + z * h[j];
      h[j] = h2;
      out[(size_t)(b * Sn + t) * SHID + j] = h2;
    }
    __syncthreads();
  }
}

// ============================================================
// Propagation GRU: hidden 64, 512 steps; mean over time -> feat[256:320]
// One block per batch, 192 threads; Wih+Whh in LDS (~60 KB, CDNA5 WGP has 320 KB).
// ============================================================
__global__ __launch_bounds__(3 * PHID)
void usr_gru_kernel(const float* __restrict__ X, const float* __restrict__ Wih,
                    const float* __restrict__ Whh, const float* __restrict__ bih,
                    const float* __restrict__ bhh, float* __restrict__ feat) {
  const int b = blockIdx.x, j = threadIdx.x;   // 192 threads
  __shared__ float Wis[3 * PHID * DIN];
  __shared__ float Whs[3 * PHID * PHID];
  __shared__ float h[PHID];
  __shared__ float gis[3 * PHID];
  __shared__ float ghs[3 * PHID];
  __shared__ float xs[DIN];
  for (int i = j; i < 3 * PHID * DIN;  i += 3 * PHID) Wis[i] = Wih[i];
  for (int i = j; i < 3 * PHID * PHID; i += 3 * PHID) Whs[i] = Whh[i];
  if (j < PHID) h[j] = 0.0f;
  const float bi = bih[j], bh = bhh[j];
  float hs = 0.0f;
  __syncthreads();
  for (int t = 0; t < Ln; ++t) {
    if (j < DIN) xs[j] = X[(size_t)(b * Ln + t) * DIN + j];
    __syncthreads();
    float gi = bi;
    const float* wi = Wis + j * DIN;
#pragma unroll
    for (int d = 0; d < DIN; ++d) gi += xs[d] * wi[d];
    float gh = bh;
    const float* wh = Whs + j * PHID;
    for (int k = 0; k < PHID; ++k) gh += h[k] * wh[k];
    gis[j] = gi; ghs[j] = gh;
    __syncthreads();
    if (j < PHID) {
      const float r = sigmoidf_(gis[j] + ghs[j]);
      const float z = sigmoidf_(gis[PHID + j] + ghs[PHID + j]);
      const float n = tanhf(gis[2 * PHID + j] + r * ghs[2 * PHID + j]);
      const float h2 = (1.0f - z) * n + z * h[j];
      h[j] = h2; hs += h2;
    }
    __syncthreads();
  }
  if (j < PHID) feat[b * FEAT + 256 + j] = hs * (1.0f / (float)Ln);
}

// ============================================================
// CNN encoder: 509 windows of 3x12 -> relu(win*filter @ Wk + bk)
// ============================================================
__global__ void cnn_kernel(const float* __restrict__ X, const float* __restrict__ filt,
                           const float* __restrict__ Wk, const float* __restrict__ bk,
                           float* __restrict__ out) {
  const int b = blockIdx.y;
  const int t = blockIdx.x * blockDim.x + threadIdx.x;
  if (t >= Pn * KSZ) return;
  const int p = t >> 6, k = t & 63;
  float acc = bk[k];
#pragma unroll
  for (int f = 0; f < FSZ; ++f)
#pragma unroll
    for (int d = 0; d < DIN; ++d)
      acc += X[(size_t)((b * Ln) + p + f) * DIN + d] * filt[f * DIN + d] *
             Wk[(f * DIN + d) * KSZ + k];
  out[((size_t)b * Pn + p) * KSZ + k] = fmaxf(acc, 0.0f);
}

// ============================================================
// Generic co-attention pieces (V: [B,S,32]; Q: stride-parameterized,
// row-stride 0 handles the broadcast gcn_out without materializing it)
// ============================================================
__global__ void co_qwb_kernel(const float* __restrict__ Q, int qbs, int qrs, int N, int qdim,
                              const float* __restrict__ Wb, float* __restrict__ qwb) {
  const int b = blockIdx.y;
  const int t = blockIdx.x * blockDim.x + threadIdx.x;
  if (t >= N * SHID) return;
  const int n = t >> 5, c = t & 31;
  const float* q = Q + (size_t)b * qbs + (size_t)n * qrs;
  float acc = 0.0f;
  for (int j = 0; j < qdim; ++j) acc += q[j] * Wb[j * SHID + c];
  qwb[((size_t)(b * N + n)) * SHID + c] = acc;
}

__global__ void co_C_kernel(const float* __restrict__ qwb, const float* __restrict__ V,
                            int N, float* __restrict__ Cm) {
  const int b = blockIdx.y;
  const int t = blockIdx.x * blockDim.x + threadIdx.x;
  if (t >= N * Sn) return;
  const int n = t >> 7, si = t & 127;
  const float* qr = qwb + ((size_t)(b * N + n)) * SHID;
  const float* vr = V + ((size_t)(b * Sn + si)) * SHID;
  float acc = 0.0f;
#pragma unroll
  for (int c = 0; c < SHID; ++c) acc += qr[c] * vr[c];
  Cm[((size_t)b * N + n) * Sn + si] = acc;
}

__global__ void co_WqQ_kernel(const float* __restrict__ Wq, const float* __restrict__ Q,
                              int qbs, int qrs, int N, int qdim, float* __restrict__ WqQ) {
  const int b = blockIdx.y;
  const int t = blockIdx.x * blockDim.x + threadIdx.x;
  if (t >= COn * N) return;
  const int c = t / N, n = t - c * N;
  const float* q = Q + (size_t)b * qbs + (size_t)n * qrs;
  float acc = 0.0f;
  for (int j = 0; j < qdim; ++j) acc += Wq[c * qdim + j] * q[j];
  WqQ[((size_t)(b * COn + c)) * N + n] = acc;
}

__global__ void co_WvV_kernel(const float* __restrict__ Wv, const float* __restrict__ V,
                              float* __restrict__ WvV) {
  const int b = blockIdx.y;
  const int t = blockIdx.x * blockDim.x + threadIdx.x;
  if (t >= COn * Sn) return;
  const int c = t >> 7, si = t & 127;
  const float* vr = V + ((size_t)(b * Sn + si)) * SHID;
  float acc = 0.0f;
#pragma unroll
  for (int j = 0; j < SHID; ++j) acc += Wv[c * SHID + j] * vr[j];
  WvV[((size_t)(b * COn + c)) * Sn + si] = acc;
}

__global__ void co_Hv_kernel(const float* __restrict__ WvV, const float* __restrict__ WqQ,
                             const float* __restrict__ Cm, int N, float* __restrict__ Hv) {
  const int b = blockIdx.y;
  const int t = blockIdx.x * blockDim.x + threadIdx.x;
  if (t >= COn * Sn) return;
  const int c = t >> 7, si = t & 127;
  float acc = WvV[((size_t)(b * COn + c)) * Sn + si];
  const float* wq = WqQ + ((size_t)(b * COn + c)) * N;
  const float* cm = Cm + (size_t)b * N * Sn + si;
  for (int n = 0; n < N; ++n) acc += wq[n] * cm[(size_t)n * Sn];
  Hv[((size_t)(b * COn + c)) * Sn + si] = tanhf(acc);
}

__global__ void co_Hq_kernel(const float* __restrict__ WqQ, const float* __restrict__ WvV,
                             const float* __restrict__ Cm, int N, float* __restrict__ Hq) {
  const int b = blockIdx.y;
  const int t = blockIdx.x * blockDim.x + threadIdx.x;
  if (t >= COn * N) return;
  const int c = t / N, n = t - c * N;
  float acc = WqQ[((size_t)(b * COn + c)) * N + n];
  const float* wv = WvV + ((size_t)(b * COn + c)) * Sn;
  const float* cm = Cm + ((size_t)(b * N + n)) * Sn;
#pragma unroll 4
  for (int s = 0; s < Sn; ++s) acc += wv[s] * cm[s];
  Hq[((size_t)(b * COn + c)) * N + n] = tanhf(acc);
}

__global__ __launch_bounds__(Sn)
void co_logit_v_kernel(const float* __restrict__ whv, const float* __restrict__ Hv,
                       float* __restrict__ lv) {
  const int b = blockIdx.x, si = threadIdx.x;
  float acc = 0.0f;
  for (int c = 0; c < COn; ++c) acc += whv[c] * Hv[((size_t)(b * COn + c)) * Sn + si];
  lv[b * Sn + si] = acc;
}

__global__ void co_logit_q_kernel(const float* __restrict__ whq, const float* __restrict__ Hq,
                                  int N, float* __restrict__ lq) {
  const int b = blockIdx.y;
  const int n = blockIdx.x * blockDim.x + threadIdx.x;
  if (n >= N) return;
  float acc = 0.0f;
  for (int c = 0; c < COn; ++c) acc += whq[c] * Hq[((size_t)(b * COn + c)) * N + n];
  lq[b * N + n] = acc;
}

// softmax over S + attended V -> feat[vOff..vOff+32)
__global__ __launch_bounds__(Sn)
void co_av_v_kernel(const float* __restrict__ lv, const float* __restrict__ V,
                    float* __restrict__ feat, int vOff) {
  const int b = blockIdx.x, t = threadIdx.x;   // 128 threads
  __shared__ float red[Sn];
  __shared__ float a[Sn];
  const float x = lv[b * Sn + t];
  red[t] = x; __syncthreads();
  for (int off = 64; off > 0; off >>= 1) {
    if (t < off) red[t] = fmaxf(red[t], red[t + off]);
    __syncthreads();
  }
  const float m = red[0]; __syncthreads();
  const float e = expf(x - m);
  red[t] = e; __syncthreads();
  for (int off = 64; off > 0; off >>= 1) {
    if (t < off) red[t] += red[t + off];
    __syncthreads();
  }
  const float inv = 1.0f / red[0]; __syncthreads();
  a[t] = e * inv; __syncthreads();
  if (t < SHID) {
    float acc = 0.0f;
    for (int s = 0; s < Sn; ++s) acc += a[s] * V[((size_t)(b * Sn + s)) * SHID + t];
    feat[b * FEAT + vOff + t] = acc;
  }
}

// softmax over N + attended Q -> feat[qOff..qOff+qdim)
__global__ __launch_bounds__(256)
void co_aq_q_kernel(const float* __restrict__ lq, const float* __restrict__ Q,
                    int qbs, int qrs, int N, int qdim,
                    float* __restrict__ feat, int qOff) {
  const int b = blockIdx.x, t = threadIdx.x;   // 256 threads, N <= 512
  __shared__ float red[256];
  __shared__ float a[512];
  const float x0 = (t < N)       ? lq[b * N + t]       : -3.4e38f;
  const float x1 = (t + 256 < N) ? lq[b * N + t + 256] : -3.4e38f;
  red[t] = fmaxf(x0, x1); __syncthreads();
  for (int off = 128; off > 0; off >>= 1) {
    if (t < off) red[t] = fmaxf(red[t], red[t + off]);
    __syncthreads();
  }
  const float m = red[0]; __syncthreads();
  const float e0 = (t < N)       ? expf(x0 - m) : 0.0f;
  const float e1 = (t + 256 < N) ? expf(x1 - m) : 0.0f;
  red[t] = e0 + e1; __syncthreads();
  for (int off = 128; off > 0; off >>= 1) {
    if (t < off) red[t] += red[t + off];
    __syncthreads();
  }
  const float inv = 1.0f / red[0]; __syncthreads();
  if (t < N)       a[t]       = e0 * inv;
  if (t + 256 < N) a[t + 256] = e1 * inv;
  __syncthreads();
  for (int qd = t; qd < qdim; qd += 256) {
    float acc = 0.0f;
    const float* qb = Q + (size_t)b * qbs + qd;
    for (int n = 0; n < N; ++n) acc += a[n] * qb[(size_t)n * qrs];
    feat[b * FEAT + qOff + qd] = acc;
  }
}

// ============================================================
// Classifier: softmax(relu(feat @ fcW^T + fcb))
// ============================================================
__global__ __launch_bounds__(Bn)
void final_kernel(const float* __restrict__ feat, const float* __restrict__ fcW,
                  const float* __restrict__ fcb, float* __restrict__ out) {
  const int b = threadIdx.x;
  if (b >= Bn) return;
  const float* f = feat + b * FEAT;
  float p0 = fcb[0], p1 = fcb[1];
  for (int i = 0; i < FEAT; ++i) {
    p0 += f[i] * fcW[i];
    p1 += f[i] * fcW[FEAT + i];
  }
  p0 = fmaxf(p0, 0.0f); p1 = fmaxf(p1, 0.0f);
  const float m = fmaxf(p0, p1);
  const float e0 = expf(p0 - m), e1 = expf(p1 - m);
  const float inv = 1.0f / (e0 + e1);
  out[b * 2 + 0] = e0 * inv;
  out[b * 2 + 1] = e1 * inv;
}

// ============================================================
// Host side
// ============================================================
static inline int cdiv_i(int a, int b) { return (a + b - 1) / b; }

static void run_coattn(const float* V, const float* Q, int qbs, int qrs, int N, int qdim,
                       const float* Wb, const float* Wv, const float* Wq,
                       const float* whv, const float* whq,
                       float* qwb, float* Cm, float* WqQ, float* WvV,
                       float* Hv, float* Hq, float* lv, float* lq,
                       float* feat, int vOff, int qOff, hipStream_t stream) {
  co_qwb_kernel<<<dim3(cdiv_i(N * SHID, 256), Bn), 256, 0, stream>>>(Q, qbs, qrs, N, qdim, Wb, qwb);
  co_C_kernel<<<dim3(cdiv_i(N * Sn, 256), Bn), 256, 0, stream>>>(qwb, V, N, Cm);
  co_WqQ_kernel<<<dim3(cdiv_i(COn * N, 256), Bn), 256, 0, stream>>>(Wq, Q, qbs, qrs, N, qdim, WqQ);
  co_WvV_kernel<<<dim3(cdiv_i(COn * Sn, 256), Bn), 256, 0, stream>>>(Wv, V, WvV);
  co_Hv_kernel<<<dim3(cdiv_i(COn * Sn, 256), Bn), 256, 0, stream>>>(WvV, WqQ, Cm, N, Hv);
  co_Hq_kernel<<<dim3(cdiv_i(COn * N, 256), Bn), 256, 0, stream>>>(WqQ, WvV, Cm, N, Hq);
  co_logit_v_kernel<<<Bn, Sn, 0, stream>>>(whv, Hv, lv);
  co_logit_q_kernel<<<dim3(cdiv_i(N, 256), Bn), 256, 0, stream>>>(whq, Hq, N, lq);
  co_av_v_kernel<<<Bn, Sn, 0, stream>>>(lv, V, feat, vOff);
  co_aq_q_kernel<<<Bn, 256, 0, stream>>>(lq, Q, qbs, qrs, N, qdim, feat, qOff);
}

extern "C" void kernel_launch(void* const* d_in, const int* in_sizes, int n_in,
                              void* d_out, int out_size, void* d_ws, size_t ws_size,
                              hipStream_t stream) {
  (void)in_sizes; (void)n_in; (void)out_size; (void)ws_size;
  const float* source_text = (const float*)d_in[0];
  const float* user_batch  = (const float*)d_in[1];
  const float* gcn_W1 = (const float*)d_in[2];
  const float* gcn_b1 = (const float*)d_in[3];
  const float* gcn_W2 = (const float*)d_in[4];
  const float* gcn_b2 = (const float*)d_in[5];
  const float* src_fc_W = (const float*)d_in[6];
  const float* src_fc_b = (const float*)d_in[7];
  const float* src_Wih  = (const float*)d_in[8];
  const float* src_Whh  = (const float*)d_in[9];
  const float* src_bih  = (const float*)d_in[10];
  const float* src_bhh  = (const float*)d_in[11];
  const float* cnn_filter = (const float*)d_in[12];
  const float* cnn_Wk     = (const float*)d_in[13];
  const float* cnn_bk     = (const float*)d_in[14];
  const float* usr_Wih = (const float*)d_in[15];
  const float* usr_Whh = (const float*)d_in[16];
  const float* usr_bih = (const float*)d_in[17];
  const float* usr_bhh = (const float*)d_in[18];
  const float* Wb1 = (const float*)d_in[19];
  const float* Wv1 = (const float*)d_in[20];
  const float* Wq1 = (const float*)d_in[21];
  const float* whv1 = (const float*)d_in[22];
  const float* whq1 = (const float*)d_in[23];
  const float* Wb2 = (const float*)d_in[24];
  const float* Wv2 = (const float*)d_in[25];
  const float* Wq2 = (const float*)d_in[26];
  const float* whv2 = (const float*)d_in[27];
  const float* whq2 = (const float*)d_in[28];
  const float* fc_W = (const float*)d_in[29];
  const float* fc_b = (const float*)d_in[30];

  // ---- workspace carve-up (floats) ----
  float* ws = (float*)d_ws;
  size_t o = 0;
  float* g    = ws + o; o += (size_t)Bn * GOUT;              // [64,128]
  float* s    = ws + o; o += (size_t)Bn * Sn * SMID;         // [8192,256]
  float* xp   = ws + o; o += (size_t)Bn * Sn * 3 * SHID;     // [8192,96]
  float* srco = ws + o; o += (size_t)Bn * Sn * SHID;         // [64,128,32]
  float* cnno = ws + o; o += (size_t)Bn * Pn * KSZ;          // [64,509,64]
  float* feat = ws + o; o += (size_t)Bn * FEAT;              // [64,320]
  float* qwb  = ws + o; o += (size_t)Bn * 512 * SHID;
  float* Cm   = ws + o; o += (size_t)Bn * 512 * Sn;
  float* WqQ  = ws + o; o += (size_t)Bn * COn * 512;
  float* WvV  = ws + o; o += (size_t)Bn * COn * Sn;
  float* Hv   = ws + o; o += (size_t)Bn * COn * Sn;
  float* Hq   = ws + o; o += (size_t)Bn * COn * 512;
  float* lv   = ws + o; o += (size_t)Bn * Sn;
  float* lq   = ws + o; o += (size_t)Bn * 512;

  // 1) GCN -> g [B,128]
  gcn_kernel<<<Bn, GHID, 0, stream>>>(user_batch, gcn_W1, gcn_b1, gcn_W2, gcn_b2, g);

  // 2) Source FC (big WMMA GEMM): s = tanh(src @ W^T + b)  M=8192,N=256,K=3347
  //    NT=4: each wave does a 16x64 strip, A fragment reused 4x.
  wmma_gemm_nt<4><<<dim3((Bn * Sn) / 16, SMID / 64), 32, 0, stream>>>(
      source_text, src_fc_W, src_fc_b, s, Bn * Sn, SMID, SIND, 1);

  // 3) GRU input projection (WMMA GEMM): xp = s @ Wih^T + bih  M=8192,N=96,K=256
  //    NT=2: 16x32 strip per wave, 96/32 = 3 strips.
  wmma_gemm_nt<2><<<dim3((Bn * Sn) / 16, (3 * SHID) / 32), 32, 0, stream>>>(
      s, src_Wih, src_bih, xp, Bn * Sn, 3 * SHID, SMID, 0);

  // 4) Source GRU scan -> srco [B,S,32]
  src_gru_kernel<<<Bn, 3 * SHID, 0, stream>>>(xp, src_Whh, src_bhh, srco);

  // 5) CNN encoder -> cnno [B,509,64]
  cnn_kernel<<<dim3(cdiv_i(Pn * KSZ, 256), Bn), 256, 0, stream>>>(
      user_batch, cnn_filter, cnn_Wk, cnn_bk, cnno);

  // 6) Propagation GRU + temporal mean -> feat[:,256:320]
  usr_gru_kernel<<<Bn, 3 * PHID, 0, stream>>>(user_batch, usr_Wih, usr_Whh,
                                              usr_bih, usr_bhh, feat);

  // 7) Co-attention 1: V=srco, Q=g broadcast (row stride 0), N=512, qdim=128
  run_coattn(srco, g, GOUT, 0, Ln, GOUT, Wb1, Wv1, Wq1, whv1, whq1,
             qwb, Cm, WqQ, WvV, Hv, Hq, lv, lq, feat, /*vOff=*/0, /*qOff=*/32, stream);

  // 8) Co-attention 2: V=srco, Q=cnno, N=509, qdim=64
  run_coattn(srco, cnno, Pn * KSZ, KSZ, Pn, KSZ, Wb2, Wv2, Wq2, whv2, whq2,
             qwb, Cm, WqQ, WvV, Hv, Hq, lv, lq, feat, /*vOff=*/160, /*qOff=*/192, stream);

  // 9) Classifier -> d_out [64,2]
  final_kernel<<<1, Bn, 0, stream>>>(feat, fc_W, fc_b, (float*)d_out);
}
